// TemporalBlock_67095979098445
// MI455X (gfx1250) — matrix operand order repaired and checked by reading, not compiled, not run on hardware
//
#include <hip/hip_runtime.h>
#include <hip/hip_bf16.h>
#include <math.h>

// ---------------- problem constants ----------------
#define Dm   1024
#define H_   16
#define DH   64
#define F_   4096
#define E_   8
#define Bb   2
#define T_   1024
#define Nt   2048          // B*T tokens
#define D3   3072          // 3*D
#define CAPT 4352          // Nt*K + E*32 padding (32-aligned expert segments)

// ---------------- WMMA types (gfx1250, wave32) ----------------
typedef __attribute__((ext_vector_type(16))) __bf16 v16bf;
typedef __attribute__((ext_vector_type(2)))  __bf16 bf16x2;
typedef __attribute__((ext_vector_type(8)))  float  v8f;

static __device__ inline unsigned short f2bf(float f) {
    unsigned int u = __builtin_bit_cast(unsigned int, f);
    u += 0x7fffu + ((u >> 16) & 1u);      // RNE
    return (unsigned short)(u >> 16);
}

// A fragment: 16x32 bf16 tile from row-major [*, lda] at (row0, k0).
// ISA 7.12.2: lane r=lane&15 (row), half=lane>>4; vgpr v holds K = (v<4?0:16)+8*half+2*(v&3) (+1)
static __device__ inline v16bf load_frag_A(const unsigned short* A, int lda,
                                           int row0, int k0, int lane) {
    const int r = row0 + (lane & 15);
    const int half = lane >> 4;
    const unsigned short* base = A + (size_t)r * lda + k0;
    v16bf a;
#pragma unroll
    for (int v = 0; v < 8; ++v) {
        const int kk = ((v & 4) ? 16 : 0) + 8 * half + 2 * (v & 3);
        unsigned int p = *(const unsigned int*)(base + kk);   // two packed bf16
        bf16x2 t = __builtin_bit_cast(bf16x2, p);
        a[2 * v] = t[0]; a[2 * v + 1] = t[1];
    }
    return a;
}

// B fragment: 32x16 bf16 tile, B given TRANSPOSED (row-major N x K, ld=ldb)
// so K-pairs are contiguous. lane c=lane&15 (col), vgpr v holds K=(v<4?0:16)+8*half+2*(v&3) (+1).
static __device__ inline v16bf load_frag_Bt(const unsigned short* Bt, int ldb,
                                            int k0, int col0, int lane) {
    const int c = col0 + (lane & 15);
    const int half = lane >> 4;
    const unsigned short* base = Bt + (size_t)c * ldb + k0;
    v16bf b;
#pragma unroll
    for (int v = 0; v < 8; ++v) {
        const int kk = ((v & 4) ? 16 : 0) + 8 * half + 2 * (v & 3);
        unsigned int p = *(const unsigned int*)(base + kk);
        bf16x2 t = __builtin_bit_cast(bf16x2, p);
        b[2 * v] = t[0]; b[2 * v + 1] = t[1];
    }
    return b;
}

// 32x64 output tile per wave: 2 A frags x 4 B frags -> 8 wmmas per 32-deep k step.
// acc[i*4+j] covers rows row0+16i.., cols col0+16j..
static __device__ inline void wmma_tile_32x64(const unsigned short* A, int lda,
                                              const unsigned short* Bt, int ldb,
                                              int row0, int col0, int Kd, int lane,
                                              v8f acc[8]) {
    for (int k0 = 0; k0 < Kd; k0 += 32) {
        if (k0 + 32 < Kd) {
            __builtin_prefetch(A  + (size_t)(row0 + (lane & 15)) * lda + k0 + 32);
            __builtin_prefetch(Bt + (size_t)(col0 + (lane & 15)) * ldb + k0 + 32);
        }
        v16bf a0 = load_frag_A(A, lda, row0,      k0, lane);
        v16bf a1 = load_frag_A(A, lda, row0 + 16, k0, lane);
#pragma unroll
        for (int j = 0; j < 4; ++j) {
            v16bf b = load_frag_Bt(Bt, ldb, k0, col0 + 16 * j, lane);
            acc[j]     = __builtin_amdgcn_wmma_f32_16x16x32_bf16(false, a0, false, b,
                                                                 (short)0, acc[j],     false, false);
            acc[4 + j] = __builtin_amdgcn_wmma_f32_16x16x32_bf16(false, a1, false, b,
                                                                 (short)0, acc[4 + j], false, false);
        }
    }
}

// ---------------- conversion kernels ----------------
__global__ __launch_bounds__(256) void k_cvt(const float* src, unsigned short* dst, int n) {
    int i = blockIdx.x * 256 + threadIdx.x;
    if (i < n) dst[i] = f2bf(src[i]);
}

// w1 (E,D,F) -> w1t (E,F,D)  [Bt layout for expert GEMM1]
__global__ __launch_bounds__(256) void k_cvt_t_w1(const float* src, unsigned short* dst) {
    size_t i = (size_t)blockIdx.x * 256 + threadIdx.x;   // over E*F*D
    int e = (int)(i >> 22);
    int r = (int)(i & 4194303);
    int f = r >> 10, d = r & 1023;
    dst[i] = f2bf(src[((size_t)e << 22) + ((size_t)d << 12) + f]);
}

// w2 (E,F,D) -> w2t (E,D,F)  [Bt layout for expert GEMM2]
__global__ __launch_bounds__(256) void k_cvt_t_w2(const float* src, unsigned short* dst) {
    size_t i = (size_t)blockIdx.x * 256 + threadIdx.x;   // over E*D*F
    int e = (int)(i >> 22);
    int r = (int)(i & 4194303);
    int d = r >> 12, f = r & 4095;
    dst[i] = f2bf(src[((size_t)e << 22) + ((size_t)f << 10) + d]);
}

// split qkv (N,3D) f32 -> q,k [bh,t,dh] bf16 and vT [bh,dh,t] bf16
__global__ __launch_bounds__(256) void k_split_qkv(const float* qkv,
        unsigned short* qb, unsigned short* kb, unsigned short* vtb) {
    int i = blockIdx.x * 256 + threadIdx.x;              // over Bb*H_*T_*DH
    int d = i & 63, t = (i >> 6) & 1023, bh = i >> 16;
    int b = bh >> 4, h = bh & 15;
    size_t n = (size_t)b * T_ + t;
    const float* row = qkv + n * D3 + h * DH + d;
    qb[((size_t)bh * T_ + t) * DH + d]  = f2bf(row[0]);
    kb[((size_t)bh * T_ + t) * DH + d]  = f2bf(row[Dm]);
    vtb[((size_t)bh * DH + d) * T_ + t] = f2bf(row[2 * Dm]);
}

// ---------------- layernorm ----------------
__global__ __launch_bounds__(256) void k_ln(const float* X, const float* g, const float* bta,
                                            unsigned short* out_bf, float* out_f) {
    int n = blockIdx.x, tid = threadIdx.x;
    const float* xr = X + (size_t)n * Dm;
    __shared__ float s1[256], s2[256];
    float a = 0.f, b = 0.f;
    for (int d = tid; d < Dm; d += 256) { float v = xr[d]; a += v; b += v * v; }
    s1[tid] = a; s2[tid] = b; __syncthreads();
    for (int st = 128; st > 0; st >>= 1) {
        if (tid < st) { s1[tid] += s1[tid + st]; s2[tid] += s2[tid + st]; }
        __syncthreads();
    }
    float mean = s1[0] * (1.0f / Dm);
    float var  = s2[0] * (1.0f / Dm) - mean * mean;
    float inv  = rsqrtf(var + 1e-5f);
    for (int d = tid; d < Dm; d += 256) {
        float v = (xr[d] - mean) * inv * g[d] + bta[d];
        out_bf[(size_t)n * Dm + d] = f2bf(v);
        if (out_f) out_f[(size_t)n * Dm + d] = v;
    }
}

// ---------------- generic single-wave 32x64 WMMA GEMM: C = A * Bt^T (+bias,+resid) ----------------
__global__ __launch_bounds__(32) void k_gemm(const unsigned short* A, int lda,
                                             const unsigned short* Bt, int ldb,
                                             const float* bias, const float* resid,
                                             float* Cf, unsigned short* Cbf,
                                             int Nn, int Kd) {
    const int lane = threadIdx.x & 31;
    const int row0 = blockIdx.x * 32, col0 = blockIdx.y * 64;
    v8f acc[8];
    v8f zed = {};
#pragma unroll
    for (int i = 0; i < 8; ++i) acc[i] = zed;
    wmma_tile_32x64(A, lda, Bt, ldb, row0, col0, Kd, lane, acc);

    const int cl = lane & 15, half = lane >> 4;
#pragma unroll
    for (int i = 0; i < 2; ++i) {
#pragma unroll
        for (int j = 0; j < 4; ++j) {
            int col = col0 + 16 * j + cl;
            float bcol = bias ? bias[col] : 0.f;
#pragma unroll
            for (int v = 0; v < 8; ++v) {
                int row = row0 + 16 * i + 8 * half + v;
                float x = acc[i * 4 + j][v] + bcol;
                if (resid) x += resid[(size_t)row * Nn + col];
                if (Cf)  Cf [(size_t)row * Nn + col] = x;
                if (Cbf) Cbf[(size_t)row * Nn + col] = f2bf(x);
            }
        }
    }
}

// ---------------- flash attention: one wave = one (b,h) x 16-query tile ----------------
__global__ __launch_bounds__(32) void k_attn(const unsigned short* qb, const unsigned short* kb,
                                             const unsigned short* vtb, unsigned short* ctx) {
    const int lane = threadIdx.x & 31;
    const int qt = blockIdx.x;            // 0..T/16-1
    const int bh = blockIdx.y;            // 0..B*H-1
    const int b = bh >> 4, h = bh & 15;
    const unsigned short* Q  = qb  + ((size_t)bh * T_ + qt * 16) * DH;
    const unsigned short* K  = kb  + (size_t)bh * T_ * DH;
    const unsigned short* VT = vtb + (size_t)bh * DH * T_;

    v16bf aq0 = load_frag_A(Q, DH, 0, 0,  lane);
    v16bf aq1 = load_frag_A(Q, DH, 0, 32, lane);

    float m[8], l[8];
    v8f O[4];
    v8f zed = {};
#pragma unroll
    for (int v = 0; v < 8; ++v) { m[v] = -1e30f; l[v] = 0.f; }
#pragma unroll
    for (int j = 0; j < 4; ++j) O[j] = zed;

    __shared__ unsigned short sP[16 * 32];

    for (int kt = 0; kt < T_; kt += 32) {
        // S = Q K^T for 32 keys: two 16x16 score tiles, K-dim 64 split into 2 wmmas each
        v8f S0 = {}, S1 = {};
        {
            v16bf kb0 = load_frag_Bt(K + (size_t)kt * DH,        DH, 0,  0, lane);
            v16bf kb1 = load_frag_Bt(K + (size_t)kt * DH,        DH, 32, 0, lane);
            S0 = __builtin_amdgcn_wmma_f32_16x16x32_bf16(false, aq0, false, kb0, (short)0, S0, false, false);
            S0 = __builtin_amdgcn_wmma_f32_16x16x32_bf16(false, aq1, false, kb1, (short)0, S0, false, false);
            v16bf kb2 = load_frag_Bt(K + (size_t)(kt + 16) * DH, DH, 0,  0, lane);
            v16bf kb3 = load_frag_Bt(K + (size_t)(kt + 16) * DH, DH, 32, 0, lane);
            S1 = __builtin_amdgcn_wmma_f32_16x16x32_bf16(false, aq0, false, kb2, (short)0, S1, false, false);
            S1 = __builtin_amdgcn_wmma_f32_16x16x32_bf16(false, aq1, false, kb3, (short)0, S1, false, false);
        }
        // online softmax (row lives across the 16 lanes of a half-wave)
        float p0[8], p1[8], fs[8];
#pragma unroll
        for (int v = 0; v < 8; ++v) {
            float s0 = S0[v] * 0.125f, s1 = S1[v] * 0.125f;
            float t = fmaxf(s0, s1);
            for (int mk = 1; mk < 16; mk <<= 1) t = fmaxf(t, __shfl_xor(t, mk, 32));
            float mn = fmaxf(m[v], t);
            float f  = __expf(m[v] - mn);
            p0[v] = __expf(s0 - mn); p1[v] = __expf(s1 - mn);
            float r = p0[v] + p1[v];
            for (int mk = 1; mk < 16; mk <<= 1) r += __shfl_xor(r, mk, 32);
            l[v] = l[v] * f + r;
            m[v] = mn; fs[v] = f;
        }
#pragma unroll
        for (int j = 0; j < 4; ++j)
#pragma unroll
            for (int v = 0; v < 8; ++v) O[j][v] *= fs[v];

        // P: C-layout -> A-layout via LDS bounce
        {
            int cb = lane & 15, half = lane >> 4;
#pragma unroll
            for (int v = 0; v < 8; ++v) {
                int row = 8 * half + v;
                sP[row * 32 + cb]      = f2bf(p0[v]);
                sP[row * 32 + 16 + cb] = f2bf(p1[v]);
            }
        }
        __syncthreads();
        v16bf aP = load_frag_A(sP, 32, 0, 0, lane);
        __syncthreads();
        // O += P (16x32) * V (32x64): 4 wmmas over 16-wide dh chunks, V stored transposed
#pragma unroll
        for (int j = 0; j < 4; ++j) {
            v16bf bv = load_frag_Bt(VT + (size_t)(j * 16) * T_ + kt, T_, 0, 0, lane);
            O[j] = __builtin_amdgcn_wmma_f32_16x16x32_bf16(false, aP, false, bv,
                                                           (short)0, O[j], false, false);
        }
    }
    // normalize and write ctx as bf16 in (B,T,D) layout
    const int col = lane & 15, half = lane >> 4;
#pragma unroll
    for (int v = 0; v < 8; ++v) {
        float inv = 1.0f / l[v];
        int t = qt * 16 + 8 * half + v;
        size_t base = ((size_t)b * T_ + t) * Dm + h * DH;
#pragma unroll
        for (int j = 0; j < 4; ++j)
            ctx[base + j * 16 + col] = f2bf(O[j][v] * inv);
    }
}

// ---------------- router / top-2 / deterministic bucketing ----------------
// meta: [0..7] counts, [8..16] offsets, [17..24] cursors
__global__ __launch_bounds__(32) void k_meta_init(int* meta) {
    if (threadIdx.x < 25) meta[threadIdx.x] = 0;
}

__global__ __launch_bounds__(256) void k_router(const float* h2f, const float* rw,
                                                int* topki, float* topkw, int* meta) {
    int n = blockIdx.x, tid = threadIdx.x;
    __shared__ float red[256];
    __shared__ float lg[E_];
    float p[E_] = {0.f};
    const float* xr = h2f + (size_t)n * Dm;
    for (int d = tid; d < Dm; d += 256) {
        float xv = xr[d];
#pragma unroll
        for (int e = 0; e < E_; ++e) p[e] += xv * rw[e * Dm + d];
    }
    for (int e = 0; e < E_; ++e) {
        red[tid] = p[e]; __syncthreads();
        for (int st = 128; st > 0; st >>= 1) {
            if (tid < st) red[tid] += red[tid + st];
            __syncthreads();
        }
        if (tid == 0) lg[e] = red[0];
        __syncthreads();
    }
    if (tid == 0) {
        float m1 = -1e30f, m2 = -1e30f; int i1 = 0, i2 = 0;
        for (int e = 0; e < E_; ++e) {
            float v = lg[e];
            if (v > m1)      { m2 = m1; i2 = i1; m1 = v; i1 = e; }
            else if (v > m2) { m2 = v; i2 = e; }
        }
        float w0 = 1.f / (1.f + __expf(m2 - m1));   // softmax over top-2
        topki[n * 2 + 0] = i1; topki[n * 2 + 1] = i2;
        topkw[n * 2 + 0] = w0; topkw[n * 2 + 1] = 1.f - w0;
        atomicAdd(&meta[i1], 1);
        atomicAdd(&meta[i2], 1);
    }
}

__global__ __launch_bounds__(32) void k_offsets(int* meta) {
    if (threadIdx.x == 0) {
        meta[8] = 0;
        for (int e = 0; e < E_; ++e)
            meta[9 + e] = meta[8 + e] + ((meta[e] + 31) & ~31);  // 32-aligned segments
    }
}

__global__ __launch_bounds__(256) void k_assign(const int* topki, int* meta, int* slotmap,
                                                const unsigned short* h2bf, unsigned short* GA) {
    int n = blockIdx.x, tid = threadIdx.x;
    __shared__ int ss[2];
    if (tid == 0) {
#pragma unroll
        for (int k = 0; k < 2; ++k) {
            int e = topki[n * 2 + k];
            int pos = atomicAdd(&meta[17 + e], 1);
            int slot = meta[8 + e] + pos;
            slotmap[n * 2 + k] = slot;
            ss[k] = slot;
        }
    }
    __syncthreads();
    for (int k = 0; k < 2; ++k) {
        int slot = ss[k];
        for (int d = tid; d < Dm; d += 256)
            GA[(size_t)slot * Dm + d] = h2bf[(size_t)n * Dm + d];
    }
}

// ---------------- expert GEMMs (segment lookup from meta offsets; 32x64 tiles) ----------------
__global__ __launch_bounds__(32) void k_moe_gemm1(const unsigned short* GA, const unsigned short* W1T,
                                                  const float* b1, unsigned short* hid,
                                                  const int* meta) {
    const int row0 = blockIdx.x * 32;
    const int* offs = meta + 8;
    if (row0 >= offs[E_]) return;
    int e = 0;
#pragma unroll
    for (int i = 0; i < E_; ++i) if (row0 >= offs[i + 1]) e = i + 1;

    const int lane = threadIdx.x & 31;
    const int col0 = blockIdx.y * 64;
    const unsigned short* Bt = W1T + (size_t)e * F_ * Dm;   // (F, D)
    v8f acc[8];
    v8f zed = {};
#pragma unroll
    for (int i = 0; i < 8; ++i) acc[i] = zed;
    wmma_tile_32x64(GA, Dm, Bt, Dm, row0, col0, Dm, lane, acc);

    const int cl = lane & 15, half = lane >> 4;
#pragma unroll
    for (int i = 0; i < 2; ++i) {
#pragma unroll
        for (int j = 0; j < 4; ++j) {
            int col = col0 + 16 * j + cl;
            float bc = b1[e * F_ + col];
#pragma unroll
            for (int v = 0; v < 8; ++v) {
                int row = row0 + 16 * i + 8 * half + v;
                float x = acc[i * 4 + j][v] + bc;
                x = 0.5f * x * (1.0f + erff(x * 0.70710678f));   // exact GELU
                hid[(size_t)row * F_ + col] = f2bf(x);
            }
        }
    }
}

__global__ __launch_bounds__(32) void k_moe_gemm2(const unsigned short* hid, const unsigned short* W2T,
                                                  const float* b2, float* eo, const int* meta) {
    const int row0 = blockIdx.x * 32;
    const int* offs = meta + 8;
    if (row0 >= offs[E_]) return;
    int e = 0;
#pragma unroll
    for (int i = 0; i < E_; ++i) if (row0 >= offs[i + 1]) e = i + 1;

    const int lane = threadIdx.x & 31;
    const int col0 = blockIdx.y * 64;
    const unsigned short* Bt = W2T + (size_t)e * Dm * F_;   // (D, F)
    v8f acc[8];
    v8f zed = {};
#pragma unroll
    for (int i = 0; i < 8; ++i) acc[i] = zed;
    wmma_tile_32x64(hid, F_, Bt, F_, row0, col0, F_, lane, acc);

    const int cl = lane & 15, half = lane >> 4;
#pragma unroll
    for (int i = 0; i < 2; ++i) {
#pragma unroll
        for (int j = 0; j < 4; ++j) {
            int col = col0 + 16 * j + cl;
            float bc = b2[e * Dm + col];
#pragma unroll
            for (int v = 0; v < 8; ++v) {
                int row = row0 + 16 * i + 8 * half + v;
                eo[(size_t)row * Dm + col] = acc[i * 4 + j][v] + bc;
            }
        }
    }
}

// ---------------- final combine ----------------
__global__ __launch_bounds__(256) void k_combine(const float* x1, const float* eo,
                                                 const int* slotmap, const float* topkw, float* out) {
    int n = blockIdx.x, tid = threadIdx.x;
    int s0 = slotmap[n * 2], s1 = slotmap[n * 2 + 1];
    float w0 = topkw[n * 2], w1 = topkw[n * 2 + 1];
    for (int d = tid; d < Dm; d += 256)
        out[(size_t)n * Dm + d] = x1[(size_t)n * Dm + d]
                                + w0 * eo[(size_t)s0 * Dm + d]
                                + w1 * eo[(size_t)s1 * Dm + d];
}

// ---------------- host side ----------------
extern "C" void kernel_launch(void* const* d_in, const int* in_sizes, int n_in,
                              void* d_out, int out_size, void* d_ws, size_t ws_size,
                              hipStream_t stream) {
    const float* x        = (const float*)d_in[0];
    const float* ln1_g    = (const float*)d_in[1];
    const float* ln1_b    = (const float*)d_in[2];
    const float* in_w     = (const float*)d_in[3];
    const float* in_b     = (const float*)d_in[4];
    const float* out_w    = (const float*)d_in[5];
    const float* out_b    = (const float*)d_in[6];
    const float* ln2_g    = (const float*)d_in[7];
    const float* ln2_b    = (const float*)d_in[8];
    const float* router_w = (const float*)d_in[9];
    const float* w1       = (const float*)d_in[10];
    const float* b1       = (const float*)d_in[11];
    const float* w2       = (const float*)d_in[12];
    const float* b2       = (const float*)d_in[13];
    float* out = (float*)d_out;

    // workspace bump allocator (256B aligned)
    char* base = (char*)d_ws;
    size_t off = 0;
    auto alloc = [&](size_t bytes) -> char* {
        char* p = base + off;
        off = (off + bytes + 255) & ~(size_t)255;
        return p;
    };
    unsigned short* inw_bf = (unsigned short*)alloc((size_t)D3 * Dm * 2);
    unsigned short* outw_bf= (unsigned short*)alloc((size_t)Dm * Dm * 2);
    unsigned short* w1t_bf = (unsigned short*)alloc((size_t)E_ * F_ * Dm * 2);
    unsigned short* w2t_bf = (unsigned short*)alloc((size_t)E_ * Dm * F_ * 2);
    unsigned short* h1_bf  = (unsigned short*)alloc((size_t)Nt * Dm * 2);
    float*          qkv_f  = (float*)alloc((size_t)Nt * D3 * 4);
    unsigned short* q_bf   = (unsigned short*)alloc((size_t)Bb * H_ * T_ * DH * 2);
    unsigned short* k_bf   = (unsigned short*)alloc((size_t)Bb * H_ * T_ * DH * 2);
    unsigned short* vt_bf  = (unsigned short*)alloc((size_t)Bb * H_ * DH * T_ * 2);
    unsigned short* ctx_bf = (unsigned short*)alloc((size_t)Nt * Dm * 2);
    float*          x1_f   = (float*)alloc((size_t)Nt * Dm * 4);
    unsigned short* h2_bf  = (unsigned short*)alloc((size_t)Nt * Dm * 2);
    float*          h2_f   = (float*)alloc((size_t)Nt * Dm * 4);
    int*            topki  = (int*)alloc((size_t)Nt * 2 * 4);
    float*          topkw  = (float*)alloc((size_t)Nt * 2 * 4);
    int*            slotmap= (int*)alloc((size_t)Nt * 2 * 4);
    int*            meta   = (int*)alloc(32 * 4);
    unsigned short* GA     = (unsigned short*)alloc((size_t)CAPT * Dm * 2);
    unsigned short* hid    = (unsigned short*)alloc((size_t)CAPT * F_ * 2);
    float*          eo     = (float*)alloc((size_t)CAPT * Dm * 4);
    (void)n_in; (void)in_sizes; (void)out_size; (void)ws_size;

    // weight precision conversion (one-time per launch; bf16 feeds WMMA)
    k_cvt<<<(D3 * Dm + 255) / 256, 256, 0, stream>>>(in_w, inw_bf, D3 * Dm);
    k_cvt<<<(Dm * Dm + 255) / 256, 256, 0, stream>>>(out_w, outw_bf, Dm * Dm);
    k_cvt_t_w1<<<(E_ * (size_t)F_ * Dm) / 256, 256, 0, stream>>>(w1, w1t_bf);
    k_cvt_t_w2<<<(E_ * (size_t)Dm * F_) / 256, 256, 0, stream>>>(w2, w2t_bf);

    // LN1 -> h1 (bf16)
    k_ln<<<Nt, 256, 0, stream>>>(x, ln1_g, ln1_b, h1_bf, nullptr);
    // qkv = h1 @ in_w^T + in_b   (32x64 tiles)
    k_gemm<<<dim3(Nt / 32, D3 / 64), 32, 0, stream>>>(h1_bf, Dm, inw_bf, Dm,
            in_b, nullptr, qkv_f, nullptr, D3, Dm);
    // head split + bf16 (V transposed)
    k_split_qkv<<<(Bb * H_ * T_ * DH) / 256, 256, 0, stream>>>(qkv_f, q_bf, k_bf, vt_bf);
    // flash attention
    k_attn<<<dim3(T_ / 16, Bb * H_), 32, 0, stream>>>(q_bf, k_bf, vt_bf, ctx_bf);
    // x1 = x + ctx @ out_w^T + out_b
    k_gemm<<<dim3(Nt / 32, Dm / 64), 32, 0, stream>>>(ctx_bf, Dm, outw_bf, Dm,
            out_b, x, x1_f, nullptr, Dm, Dm);
    // LN2 -> h2 (bf16 + f32 for router)
    k_ln<<<Nt, 256, 0, stream>>>(x1_f, ln2_g, ln2_b, h2_bf, h2_f);
    // routing: top-2, deterministic slot assignment, token gather
    k_meta_init<<<1, 32, 0, stream>>>(meta);
    k_router<<<Nt, 256, 0, stream>>>(h2_f, router_w, topki, topkw, meta);
    k_offsets<<<1, 32, 0, stream>>>(meta);
    k_assign<<<Nt, 256, 0, stream>>>(topki, meta, slotmap, h2_bf, GA);
    // expert FFN (only routed tokens; 32-aligned segments so 32-row tiles never span experts)
    k_moe_gemm1<<<dim3(CAPT / 32, F_ / 64), 32, 0, stream>>>(GA, w1t_bf, b1, hid, meta);
    k_moe_gemm2<<<dim3(CAPT / 32, Dm / 64), 32, 0, stream>>>(hid, w2t_bf, b2, eo, meta);
    // out = x1 + sum_k w_k * expert_out
    k_combine<<<Nt, 256, 0, stream>>>(x1_f, eo, slotmap, topkw, out);
}
